// MultiHeadQGFDLayer_15857019257589
// MI455X (gfx1250) — matrix operand order repaired and checked by reading, not compiled, not run on hardware
//
#include <hip/hip_runtime.h>
#include <stdint.h>
#include <stddef.h>

// ---------------------------------------------------------------------------
// MultiHeadQGFD layer for MI455X (gfx1250), wave32 + WMMA bf16.
//   B=1, L=2048, D=1024, H=16, hd=64, 4 diffusion steps, alpha=0.02.
// All large GEMMs use v_wmma_f32_16x16x32_bf16 with the convention
//   C[M][N] = alpha * (A[M][K] * Bt[N][K]^T) + bias[N] + beta * blend[M][N]
// so B-fragments are contiguous 32B per lane (k = 16*(lane/16) + s, same order as A).
// Wave tile 32x64 (2 M-tiles x 4 N-tiles), block tile 64x256, 8 waves.
// Inner loop is software-pipelined (double-buffered fragments) and contains
// NO divergent control flow, so WMMA always executes with EXEC all-1s.
// ---------------------------------------------------------------------------

#define L_SEQ   2048
#define D_EMB   1024
#define N_HEADS 16
#define HD      64

typedef __attribute__((ext_vector_type(16))) __bf16 v16bf;
typedef __attribute__((ext_vector_type(8)))  __bf16 v8bf;
typedef __attribute__((ext_vector_type(8)))  float  v8f;

static __device__ __forceinline__ float bf2f(__bf16 b) {
  unsigned short h; __builtin_memcpy(&h, &b, 2);
  unsigned u = ((unsigned)h) << 16;
  float f; __builtin_memcpy(&f, &u, 4);
  return f;
}
static __device__ __forceinline__ __bf16 f2bf(float f) {
  unsigned u; __builtin_memcpy(&u, &f, 4);
  u += 0x7FFFu + ((u >> 16) & 1u);           // round-to-nearest-even
  unsigned short h = (unsigned short)(u >> 16);
  __bf16 b; __builtin_memcpy(&b, &h, 2);
  return b;
}
static __device__ __forceinline__ void storeOut(float*  p, float v) { *p = v; }
static __device__ __forceinline__ void storeOut(__bf16* p, float v) { *p = f2bf(v); }

// Assemble a 16-element bf16 fragment from two 16B-aligned 8-element loads.
static __device__ __forceinline__ v16bf load_frag(const __bf16* p0, const __bf16* p1) {
  v8bf lo = *(const v8bf*)p0;
  v8bf hi = *(const v8bf*)p1;
  v16bf r;
#pragma unroll
  for (int i = 0; i < 8; ++i) { r[i] = lo[i]; r[8 + i] = hi[i]; }
  return r;
}

// ---------------------------------------------------------------------------
// Generic bf16 WMMA GEMM.
//   grid.x = ceil(N/256), grid.y = M/64, grid.z = batch
// ---------------------------------------------------------------------------
template <typename OutT, bool TRANS_OUT, bool HAS_BLEND>
__global__ __launch_bounds__(256)
void gemm_bf16_wmma(const __bf16* __restrict__ A,   // [M][K]  (lda)
                    const __bf16* __restrict__ Bt,  // [N][K]  (ldb)  == B^T
                    OutT*         __restrict__ C,   // [M][N]  (ldc) or [N][M] if TRANS_OUT
                    const __bf16* __restrict__ blend,   // [M][N] (ldc) or null
                    const float*  __restrict__ bias,    // [N] or null
                    int M, int N, int K, int lda, int ldb, int ldc,
                    float alpha, float beta,
                    long sA, long sB, long sC, long sBlend)
{
  const int z = blockIdx.z;
  A  += (long)z * sA;
  Bt += (long)z * sB;
  C  += (long)z * sC;
  if (HAS_BLEND) blend += (long)z * sBlend;

  const int wave  = threadIdx.x >> 5;
  const int lane  = threadIdx.x & 31;
  const int waveM = wave & 1;                  // 2 waves along M
  const int waveN = wave >> 1;                 // 4 waves along N
  const int m0    = blockIdx.y * 64 + waveM * 32;   // rows m0..m0+31
  const int nBase = blockIdx.x * 256 + waveN * 64;  // 4 n-tiles of 16
  if (m0 >= M || nBase >= N) return;           // wave-uniform early-out

  const int laneHalf = lane >> 4;              // 0/1
  const int laneLo   = lane & 15;

  // A fragment base pointers for the two 16-row M-tiles.
  const __bf16* aRow0 = A + (long)(m0 + laneLo) * lda + laneHalf * 16;
  const __bf16* aRow1 = aRow0 + (long)16 * lda;
  // B fragment base pointers for the four 16-col N-tiles (contiguous in K).
  const __bf16* bPtr0 = Bt + (long)(nBase +  0 + laneLo) * ldb + laneHalf * 16;
  const __bf16* bPtr1 = Bt + (long)(nBase + 16 + laneLo) * ldb + laneHalf * 16;
  const __bf16* bPtr2 = Bt + (long)(nBase + 32 + laneLo) * ldb + laneHalf * 16;
  const __bf16* bPtr3 = Bt + (long)(nBase + 48 + laneLo) * ldb + laneHalf * 16;

  v8f acc0[4] = {};
  v8f acc1[4] = {};

  // Prologue: fragments for k = 0.
  v16bf a0 = load_frag(aRow0, aRow0 + 8);
  v16bf a1 = load_frag(aRow1, aRow1 + 8);
  v16bf b0 = load_frag(bPtr0, bPtr0 + 8);
  v16bf b1 = load_frag(bPtr1, bPtr1 + 8);
  v16bf b2 = load_frag(bPtr2, bPtr2 + 8);
  v16bf b3 = load_frag(bPtr3, bPtr3 + 8);

  for (int k0 = 32; k0 < K; k0 += 32) {
    // Issue next-step loads first so they overlap the 8 WMMAs below.
    v16bf a0n = load_frag(aRow0 + k0, aRow0 + k0 + 8);
    v16bf a1n = load_frag(aRow1 + k0, aRow1 + k0 + 8);
    v16bf b0n = load_frag(bPtr0 + k0, bPtr0 + k0 + 8);
    v16bf b1n = load_frag(bPtr1 + k0, bPtr1 + k0 + 8);
    v16bf b2n = load_frag(bPtr2 + k0, bPtr2 + k0 + 8);
    v16bf b3n = load_frag(bPtr3 + k0, bPtr3 + k0 + 8);
    __builtin_prefetch(aRow0 + k0 + 256, 0, 1);   // global_prefetch_b8

    acc0[0] = __builtin_amdgcn_wmma_f32_16x16x32_bf16(false, a0, false, b0, (short)0, acc0[0], false, false);
    acc0[1] = __builtin_amdgcn_wmma_f32_16x16x32_bf16(false, a0, false, b1, (short)0, acc0[1], false, false);
    acc0[2] = __builtin_amdgcn_wmma_f32_16x16x32_bf16(false, a0, false, b2, (short)0, acc0[2], false, false);
    acc0[3] = __builtin_amdgcn_wmma_f32_16x16x32_bf16(false, a0, false, b3, (short)0, acc0[3], false, false);
    acc1[0] = __builtin_amdgcn_wmma_f32_16x16x32_bf16(false, a1, false, b0, (short)0, acc1[0], false, false);
    acc1[1] = __builtin_amdgcn_wmma_f32_16x16x32_bf16(false, a1, false, b1, (short)0, acc1[1], false, false);
    acc1[2] = __builtin_amdgcn_wmma_f32_16x16x32_bf16(false, a1, false, b2, (short)0, acc1[2], false, false);
    acc1[3] = __builtin_amdgcn_wmma_f32_16x16x32_bf16(false, a1, false, b3, (short)0, acc1[3], false, false);

    a0 = a0n; a1 = a1n;
    b0 = b0n; b1 = b1n; b2 = b2n; b3 = b3n;
  }
  // Epilogue k-step.
  acc0[0] = __builtin_amdgcn_wmma_f32_16x16x32_bf16(false, a0, false, b0, (short)0, acc0[0], false, false);
  acc0[1] = __builtin_amdgcn_wmma_f32_16x16x32_bf16(false, a0, false, b1, (short)0, acc0[1], false, false);
  acc0[2] = __builtin_amdgcn_wmma_f32_16x16x32_bf16(false, a0, false, b2, (short)0, acc0[2], false, false);
  acc0[3] = __builtin_amdgcn_wmma_f32_16x16x32_bf16(false, a0, false, b3, (short)0, acc0[3], false, false);
  acc1[0] = __builtin_amdgcn_wmma_f32_16x16x32_bf16(false, a1, false, b0, (short)0, acc1[0], false, false);
  acc1[1] = __builtin_amdgcn_wmma_f32_16x16x32_bf16(false, a1, false, b1, (short)0, acc1[1], false, false);
  acc1[2] = __builtin_amdgcn_wmma_f32_16x16x32_bf16(false, a1, false, b2, (short)0, acc1[2], false, false);
  acc1[3] = __builtin_amdgcn_wmma_f32_16x16x32_bf16(false, a1, false, b3, (short)0, acc1[3], false, false);

  // ---- epilogue: C/D layout: lane half selects M block of 8, VGPR r = row
#pragma unroll
  for (int g = 0; g < 2; ++g) {
    const int rowBase = m0 + g * 16 + laneHalf * 8;
#pragma unroll
    for (int t = 0; t < 4; ++t) {
      const int col = nBase + t * 16 + laneLo;
      const float bb = bias ? bias[col] : 0.0f;
      const v8f acc = g ? acc1[t] : acc0[t];
#pragma unroll
      for (int r = 0; r < 8; ++r) {
        const int rr = rowBase + r;
        float v = alpha * acc[r] + bb;
        if (HAS_BLEND) v += beta * bf2f(blend[(long)rr * ldc + col]);
        if (TRANS_OUT) storeOut(&C[(long)col * ldc + rr], v);
        else           storeOut(&C[(long)rr * ldc + col], v);
      }
    }
  }
}

// ---------------------------------------------------------------------------
// fp32 -> bf16 elementwise convert
// ---------------------------------------------------------------------------
__global__ __launch_bounds__(256)
void convert_f32_bf16(const float* __restrict__ in, __bf16* __restrict__ out, long n)
{
  long i = (long)blockIdx.x * 256 + threadIdx.x;
  if (i < n) out[i] = f2bf(in[i]);
}

// ---------------------------------------------------------------------------
// fp32 W[rows][cols] -> bf16 Wt[cols][rows]   (32x32 LDS tiles)
// ---------------------------------------------------------------------------
__global__ __launch_bounds__(256)
void transpose_conv(const float* __restrict__ W, __bf16* __restrict__ Wt,
                    int rows, int cols)
{
  __shared__ float tile[32][33];
  const int c0 = blockIdx.x * 32;
  const int r0 = blockIdx.y * 32;
  for (int i = threadIdx.y; i < 32; i += 8)
    tile[i][threadIdx.x] = W[(long)(r0 + i) * cols + c0 + threadIdx.x];
  __syncthreads();
  for (int i = threadIdx.y; i < 32; i += 8)
    Wt[(long)(c0 + i) * rows + r0 + threadIdx.x] = f2bf(tile[threadIdx.x][i]);
}

// ---------------------------------------------------------------------------
// Per-(row,head) K normalization: Kn = K / max(||K||, 1e-6). One wave / row.
// ---------------------------------------------------------------------------
__global__ __launch_bounds__(256)
void knorm_kernel(const __bf16* __restrict__ K, __bf16* __restrict__ Kn)
{
  const int lane   = threadIdx.x & 31;
  const int waveId = blockIdx.x * 8 + (threadIdx.x >> 5);
  const int l = waveId >> 4;        // sequence position
  const int h = waveId & 15;        // head
  const long base = (long)l * D_EMB + h * HD + lane * 2;
  const float a = bf2f(K[base]);
  const float b = bf2f(K[base + 1]);
  float s = a * a + b * b;
#pragma unroll
  for (int d = 16; d >= 1; d >>= 1) s += __shfl_xor(s, d, 32);
  const float inv = 1.0f / fmaxf(sqrtf(s), 1e-6f);
  Kn[base]     = f2bf(a * inv);
  Kn[base + 1] = f2bf(b * inv);
}

// ---------------------------------------------------------------------------
// Row softmax over fp32 scores [H*L rows of length L=2048].
//   out = softmax(row) * mulv + addv   (bf16)
// TRANS=false: write out[row][*] (and duplicate into out2 if non-null)
// TRANS=true : write out transposed within each head:  out[h][m][r]
// ---------------------------------------------------------------------------
template <bool TRANS>
__global__ __launch_bounds__(256)
void softmax_rows(const float* __restrict__ S,
                  __bf16* __restrict__ out, __bf16* __restrict__ out2,
                  float mulv, float addv)
{
  const long rowId = blockIdx.x;                 // 0 .. H*L-1
  const float* srow = S + rowId * (long)L_SEQ;
  const int tid = threadIdx.x;

  float vals[8];
  float mx = -3.4e38f;
#pragma unroll
  for (int i = 0; i < 8; ++i) {
    vals[i] = srow[tid + i * 256];
    mx = fmaxf(mx, vals[i]);
  }
  __shared__ float red[256];
  red[tid] = mx; __syncthreads();
  for (int s = 128; s > 0; s >>= 1) {
    if (tid < s) red[tid] = fmaxf(red[tid], red[tid + s]);
    __syncthreads();
  }
  mx = red[0]; __syncthreads();

  float sum = 0.0f;
#pragma unroll
  for (int i = 0; i < 8; ++i) { vals[i] = __expf(vals[i] - mx); sum += vals[i]; }
  red[tid] = sum; __syncthreads();
  for (int s = 128; s > 0; s >>= 1) {
    if (tid < s) red[tid] += red[tid + s];
    __syncthreads();
  }
  const float inv = 1.0f / red[0];

  const long head = rowId / L_SEQ;
  const long r    = rowId % L_SEQ;
#pragma unroll
  for (int i = 0; i < 8; ++i) {
    const int m = tid + i * 256;
    const __bf16 p = f2bf(vals[i] * inv * mulv + addv);
    if (TRANS) {
      out[(head * L_SEQ + m) * (long)L_SEQ + r] = p;   // P^T per head
    } else {
      out[rowId * (long)L_SEQ + m] = p;
      if (out2) out2[rowId * (long)L_SEQ + m] = p;
    }
  }
}

// ---------------------------------------------------------------------------
// Host-side orchestration
// ---------------------------------------------------------------------------
extern "C" void kernel_launch(void* const* d_in, const int* in_sizes, int n_in,
                              void* d_out, int out_size, void* d_ws, size_t ws_size,
                              hipStream_t stream)
{
  (void)in_sizes; (void)n_in; (void)out_size; (void)ws_size;

  const float* X  = (const float*)d_in[0];
  const float* Wq = (const float*)d_in[1];
  const float* bq = (const float*)d_in[2];
  const float* Wk = (const float*)d_in[3];
  const float* bk = (const float*)d_in[4];
  const float* Wv = (const float*)d_in[5];
  const float* bv = (const float*)d_in[6];
  const float* Wo = (const float*)d_in[7];
  const float* bo = (const float*)d_in[8];
  float* Out = (float*)d_out;

  const long LD  = (long)L_SEQ * D_EMB;            // 2M elems
  const long DD  = (long)D_EMB * D_EMB;            // 1M elems
  const long HLL = (long)N_HEADS * L_SEQ * L_SEQ;  // 64M elems

  char* ws = (char*)d_ws;
  size_t off = 0;
  auto take = [&](size_t bytes) -> char* {
    char* p = ws + off;
    off = (off + bytes + 255) & ~(size_t)255;
    return p;
  };

  __bf16* Xb   = (__bf16*)take(LD * 2);       // X in bf16           [L][D]
  __bf16* Qb   = (__bf16*)take(LD * 2);       // Q                   [L][D]
  __bf16* Kb   = (__bf16*)take(LD * 2);       // K                   [L][D]
  __bf16* Knb  = (__bf16*)take(LD * 2);       // normalized K        [L][D]
  __bf16* Vtb  = (__bf16*)take(LD * 2);       // V transposed        [D][L]
  __bf16* Wt   = (__bf16*)take(DD * 2);       // current W^T (reused)[D][D]
  __bf16* Ob   = (__bf16*)take(LD * 2);       // attention output    [L][D]
  float*  Sbuf = (float*) take(HLL * 4);      // fp32 score scratch  [H][L][L]
  __bf16* P0b  = (__bf16*)take(HLL * 2);      // p0                  [H][L][L]
  __bf16* Ptb  = (__bf16*)take(HLL * 2);      // P^T                 [H][L][L]
  __bf16* pAb  = (__bf16*)take(HLL * 2);      // diffusion ping
  __bf16* pBb  = (__bf16*)take(HLL * 2);      // diffusion pong

  const float inv_sqrt_hd = 0.125f;           // 1/sqrt(64)
  const float ALPHA = 0.02f;

  // 1) X -> bf16
  convert_f32_bf16<<<(unsigned)((LD + 255) / 256), 256, 0, stream>>>(X, Xb, LD);

  // 2) Projections: Y = X @ W + b  via  X * (W^T)^T
  const dim3 tposeGrid(D_EMB / 32, D_EMB / 32), tposeBlk(32, 8);
  const dim3 projGrid(D_EMB / 256, L_SEQ / 64, 1);

  transpose_conv<<<tposeGrid, tposeBlk, 0, stream>>>(Wq, Wt, D_EMB, D_EMB);
  gemm_bf16_wmma<__bf16, false, false><<<projGrid, 256, 0, stream>>>(
      Xb, Wt, Qb, nullptr, bq, L_SEQ, D_EMB, D_EMB, D_EMB, D_EMB, D_EMB,
      1.0f, 0.0f, 0, 0, 0, 0);

  transpose_conv<<<tposeGrid, tposeBlk, 0, stream>>>(Wk, Wt, D_EMB, D_EMB);
  gemm_bf16_wmma<__bf16, false, false><<<projGrid, 256, 0, stream>>>(
      Xb, Wt, Kb, nullptr, bk, L_SEQ, D_EMB, D_EMB, D_EMB, D_EMB, D_EMB,
      1.0f, 0.0f, 0, 0, 0, 0);

  transpose_conv<<<tposeGrid, tposeBlk, 0, stream>>>(Wv, Wt, D_EMB, D_EMB);
  gemm_bf16_wmma<__bf16, true, false><<<projGrid, 256, 0, stream>>>(
      Xb, Wt, Vtb, nullptr, bv, L_SEQ, D_EMB, D_EMB, D_EMB, D_EMB, /*ldc=M*/L_SEQ,
      1.0f, 0.0f, 0, 0, 0, 0);

  // 3) K row-normalization (per head)
  knorm_kernel<<<(L_SEQ * N_HEADS) / 8, 256, 0, stream>>>(Kb, Knb);

  // 4) scores = Q K^T * 1/sqrt(hd)   (per head; Q/K strided views of [L][D])
  const dim3 attnGrid(L_SEQ / 256, L_SEQ / 64, N_HEADS);
  gemm_bf16_wmma<float, false, false><<<attnGrid, 256, 0, stream>>>(
      Qb, Kb, Sbuf, nullptr, nullptr, L_SEQ, L_SEQ, HD, D_EMB, D_EMB, L_SEQ,
      inv_sqrt_hd, 0.0f, /*sA=*/HD, /*sB=*/HD, /*sC=*/(long)L_SEQ * L_SEQ, 0);

  // 5) p0 = softmax(scores); copy into pA
  softmax_rows<false><<<N_HEADS * L_SEQ, 256, 0, stream>>>(
      Sbuf, P0b, pAb, 1.0f, 0.0f);

  // 6) sim = Kn Kn^T * 1/sqrt(hd)   (reuse Sbuf)
  gemm_bf16_wmma<float, false, false><<<attnGrid, 256, 0, stream>>>(
      Knb, Knb, Sbuf, nullptr, nullptr, L_SEQ, L_SEQ, HD, D_EMB, D_EMB, L_SEQ,
      inv_sqrt_hd, 0.0f, HD, HD, (long)L_SEQ * L_SEQ, 0);

  // 7) P = softmax(sim)*(1-1e-6) + 1e-6/L, stored transposed per head
  softmax_rows<true><<<N_HEADS * L_SEQ, 256, 0, stream>>>(
      Sbuf, Ptb, nullptr, 1.0f - 1e-6f, 1e-6f / (float)L_SEQ);

  // 8) diffusion: p <- (1-a) p0 + a (p @ P), 4 iterations, ping-pong
  __bf16* pin = pAb;
  __bf16* pout = pBb;
  for (int it = 0; it < 4; ++it) {
    gemm_bf16_wmma<__bf16, false, true><<<attnGrid, 256, 0, stream>>>(
        pin, Ptb, pout, P0b, nullptr, L_SEQ, L_SEQ, L_SEQ, L_SEQ, L_SEQ, L_SEQ,
        ALPHA, 1.0f - ALPHA,
        (long)L_SEQ * L_SEQ, (long)L_SEQ * L_SEQ,
        (long)L_SEQ * L_SEQ, (long)L_SEQ * L_SEQ);
    __bf16* t = pin; pin = pout; pout = t;
  }
  // after 4 swaps, final p is in `pin` (== pAb)

  // 9) O[h] = p[h] @ V[h]  -> Ob[L][D] (column block = head)
  const dim3 pvGrid(1, L_SEQ / 64, N_HEADS);
  gemm_bf16_wmma<__bf16, false, false><<<pvGrid, 256, 0, stream>>>(
      pin, Vtb, Ob, nullptr, nullptr, L_SEQ, HD, L_SEQ, L_SEQ, L_SEQ, D_EMB,
      1.0f, 0.0f, (long)L_SEQ * L_SEQ, (long)HD * L_SEQ, /*sC=*/HD, 0);

  // 10) out = O @ Wo + bo  (fp32 result)
  transpose_conv<<<tposeGrid, tposeBlk, 0, stream>>>(Wo, Wt, D_EMB, D_EMB);
  gemm_bf16_wmma<float, false, false><<<projGrid, 256, 0, stream>>>(
      Ob, Wt, Out, nullptr, bo, L_SEQ, D_EMB, D_EMB, D_EMB, D_EMB, D_EMB,
      1.0f, 0.0f, 0, 0, 0, 0);
}
